// Attention_89833535963384
// MI455X (gfx1250) — matrix operand order repaired and checked by reading, not compiled, not run on hardware
//
#include <hip/hip_runtime.h>

typedef __bf16 bf16;
typedef __attribute__((ext_vector_type(8)))  __bf16 bf16x8;
typedef __attribute__((ext_vector_type(16))) __bf16 bf16x16;
typedef __attribute__((ext_vector_type(8)))  float  f32x8;

static constexpr int Bsz = 2, S = 2048, E = 1024, H = 16, HD = 64;
static constexpr int Mrows = Bsz * S;   // 4096
static constexpr int GK = E;            // 1024 (K of all projection GEMMs)
static constexpr int GN = E;            // 1024 (N of all projection GEMMs)

// ---------------------------------------------------------------- helpers
static __device__ __forceinline__ bf16 f2b(float f) { return (bf16)f; } // native cvt (RNE)

// pack two elements into one DWORD for wide LDS stores
static __device__ __forceinline__ unsigned pack2(float a, float b) {
  unsigned short lo = __builtin_bit_cast(unsigned short, f2b(a));
  unsigned short hi = __builtin_bit_cast(unsigned short, f2b(b));
  return ((unsigned)hi << 16) | (unsigned)lo;
}
static __device__ __forceinline__ unsigned pack2(unsigned short a, unsigned short b) {
  return ((unsigned)b << 16) | (unsigned)a;
}
static __device__ __forceinline__ void store_o(float* p, float v) { *p = v; }
static __device__ __forceinline__ void store_o(unsigned short* p, float v) {
  *p = __builtin_bit_cast(unsigned short, f2b(v));
}
// A-fragment: lane holds M=lane%16; K = half*8 + [0..7]  and  16 + half*8 + [0..7]
static __device__ __forceinline__ bf16x16 ld_frag(const bf16* p0, const bf16* p1) {
  bf16x16 r;
  *(bf16x8*)&r       = *(const bf16x8*)p0;
  *((bf16x8*)&r + 1) = *(const bf16x8*)p1;
  return r;
}
#define WMMA_BF16(A, Bm, Cm) \
  __builtin_amdgcn_wmma_f32_16x16x32_bf16(false, (A), false, (Bm), (short)0, (Cm), false, false)

// ---------------------------------------------------------------- GEMM: Out = oscale * (X @ W^T)
// X: [4096, 1024] (float or bf16-as-ushort), W: [1024, 1024] fp32 row-major,
// Out: [4096, 1024] (bf16-as-ushort or float).
// Block: 256 threads = 8 waves; block tile 128(M) x 64(N); wave tile 16 x 64.
template <typename XT, typename OT>
__global__ void __launch_bounds__(256)
gemm_wmma(const XT* __restrict__ X, const float* __restrict__ W, OT* __restrict__ Out,
          float oscale) {
  __shared__ __align__(32) bf16 lX[128 * 32];
  __shared__ __align__(32) bf16 lW[64 * 32];
  const int tid  = threadIdx.x;
  const int wave = tid >> 5;
  const int lane = tid & 31;
  const int l16  = lane & 15;
  const int hl   = lane >> 4;
  const int mblk = blockIdx.x * 128;
  const int nblk = blockIdx.y * 64;
  const int wrow = wave * 16;

  f32x8 acc[4];
#pragma unroll
  for (int n = 0; n < 4; ++n) acc[n] = f32x8{};

  // per-thread staging sources (advance by 32 K per iter)
  const int xr = tid >> 1, xcb = (tid & 1) * 16;
  const int wr = tid >> 2, wcb = (tid & 3) * 8;
  const XT*    xsrc = X + (size_t)(mblk + xr) * GK + xcb;
  const float* wsrc = W + (size_t)(nblk + wr) * GK + wcb;
  unsigned* xdst = (unsigned*)(lX + xr * 32 + xcb);
  unsigned* wdst = (unsigned*)(lW + wr * 32 + wcb);

  for (int kb = 0; kb < GK; kb += 32) {
    if (kb + 32 < GK) {  // overlap next tile's L2->WGP fill with this tile's WMMAs
      __builtin_prefetch(xsrc + kb + 32, 0, 1);   // -> global_prefetch_b8
      __builtin_prefetch(wsrc + kb + 32, 0, 1);
    }
#pragma unroll
    for (int j = 0; j < 8; ++j)  // X tile 128x32 -> bf16 LDS (b32 packed stores)
      xdst[j] = pack2(xsrc[kb + 2 * j], xsrc[kb + 2 * j + 1]);
#pragma unroll
    for (int j = 0; j < 4; ++j)  // W tile 64x32 -> bf16 LDS
      wdst[j] = pack2(wsrc[kb + 2 * j], wsrc[kb + 2 * j + 1]);
    __syncthreads();

    const bf16* ap = lX + (wrow + l16) * 32 + hl * 8;
    bf16x16 a = ld_frag(ap, ap + 16);
#pragma unroll
    for (int n = 0; n < 4; ++n) {
      // B operand = W^T; rows-of-transpose are W rows: K = half*8 + [0..7] and 16 + half*8 + [0..7]
      const bf16* bp = lW + (n * 16 + l16) * 32 + hl * 8;
      bf16x16 b = ld_frag(bp, bp + 16);
      acc[n] = WMMA_BF16(a, b, acc[n]);
    }
    __syncthreads();
  }

#pragma unroll
  for (int n = 0; n < 4; ++n)
#pragma unroll
    for (int r = 0; r < 8; ++r) {
      const int row = mblk + wrow + r + hl * 8;
      const int col = nblk + n * 16 + l16;
      store_o(Out + (size_t)row * GN + col, acc[n][r] * oscale);
    }
}

// ---------------------------------------------------------------- flash attention
// One 32-key tile: scores (4 WMMA) -> [mask] -> online softmax -> P repack -> PV (4 WMMA).
// MASKED is instantiated only for the single boundary tile straddling the diagonal.
template <bool MASKED>
static __device__ __forceinline__ void
attn_tile(const int kb, const int qbase, const int l16, const int hl,
          const bf16* __restrict__ Kh, const bf16* __restrict__ Vh, bf16* pl,
          const bf16x16 (&aq)[2], f32x8 (&O)[4], float (&m)[8], float (&l)[8]) {
  // ---- S = Q @ K^T for 32 keys: 16x32 scores as two 16x16 f32 fragments
  f32x8 s[2];
  s[0] = f32x8{}; s[1] = f32x8{};
#pragma unroll
  for (int c = 0; c < 2; ++c)
#pragma unroll
    for (int j = 0; j < 2; ++j) {
      // B = K^T; rows-of-transpose are K rows (d contiguous): two 16B loads, d = half*8 + [0..7] and 16 + half*8 + [0..7]
      const bf16* kp0 = Kh + (size_t)(kb + j * 16 + l16) * HD + c * 32 + hl * 8;
      bf16x16 bk = ld_frag(kp0, kp0 + 16);
      s[j] = WMMA_BF16(aq[c], bk, s[j]);
    }

  // ---- causal mask, boundary tile only (C layout: row = qbase+r+hl*8, col = kb+16j+l16)
  if (MASKED) {
#pragma unroll
    for (int j = 0; j < 2; ++j)
#pragma unroll
      for (int r = 0; r < 8; ++r) {
        const int col = kb + j * 16 + l16;
        const int row = qbase + r + hl * 8;
        if (col > row) s[j][r] = -1e30f;
      }
  }

  // ---- online softmax: each row lives in 16 lanes of one half-wave
  float alpha[8];
#pragma unroll
  for (int r = 0; r < 8; ++r) {
    float rm = fmaxf(s[0][r], s[1][r]);
    rm = fmaxf(rm, __shfl_xor(rm, 1));
    rm = fmaxf(rm, __shfl_xor(rm, 2));
    rm = fmaxf(rm, __shfl_xor(rm, 4));
    rm = fmaxf(rm, __shfl_xor(rm, 8));
    const float mn = fmaxf(m[r], rm);
    alpha[r] = __expf(m[r] - mn);
    m[r] = mn;
    const float p0 = __expf(s[0][r] - mn);
    const float p1 = __expf(s[1][r] - mn);
    s[0][r] = p0; s[1][r] = p1;
    float rs = p0 + p1;
    rs += __shfl_xor(rs, 1);
    rs += __shfl_xor(rs, 2);
    rs += __shfl_xor(rs, 4);
    rs += __shfl_xor(rs, 8);
    l[r] = l[r] * alpha[r] + rs;
  }
#pragma unroll
  for (int n = 0; n < 4; ++n)
#pragma unroll
    for (int r = 0; r < 8; ++r) O[n][r] = O[n][r] * alpha[r];

  // ---- repack P (C layout) -> A layout through per-wave LDS
#pragma unroll
  for (int j = 0; j < 2; ++j)
#pragma unroll
    for (int r = 0; r < 8; ++r)
      pl[(r + hl * 8) * 32 + j * 16 + l16] = f2b(s[j][r]);
  asm volatile("s_wait_dscnt 0" ::: "memory");   // LDS RAW fence (same-wave, in-order)
  const bf16* pp = pl + l16 * 32 + hl * 8;
  bf16x16 aP = ld_frag(pp, pp + 16);

  // ---- O += P @ V  (B = V, rows-of-transpose are V columns: strided bf16 loads;
  //      per-lane base is loop-invariant, row offsets fold into the 24-bit ioffset)
#pragma unroll
  for (int n = 0; n < 4; ++n) {
    const bf16* vp0 = Vh + (size_t)(kb + hl * 8) * HD + n * 16 + l16;
    bf16x16 bv;
#pragma unroll
    for (int i = 0; i < 16; ++i) bv[i] = vp0[(i + (i >> 3) * 8) * HD];
    O[n] = WMMA_BF16(aP, bv, O[n]);
  }
}

// Per the reference's reshape(-1,H,S,HD) (no transpose), each (b,h) head is a
// CONTIGUOUS [S, HD] chunk at offset head*S*HD of the flat projection buffer.
// Block: 128 threads = 4 waves; each wave owns 16 query rows; key tiles of 32.
// NOTE: scores arrive pre-scaled (1/sqrt(HD) folded into the Q projection GEMM).
__global__ void __launch_bounds__(128)
flash_attn(const unsigned short* __restrict__ Qv, const unsigned short* __restrict__ Kv,
           const unsigned short* __restrict__ Vv, unsigned short* __restrict__ Ov,
           const int* __restrict__ maskp) {
  __shared__ __align__(32) bf16 lP[4 * 512];   // per-wave 16x32 P repack buffer
  const int tid  = threadIdx.x;
  const int wave = tid >> 5;
  const int lane = tid & 31;
  const int l16  = lane & 15;
  const int hl   = lane >> 4;
  const int qbase = (blockIdx.x * 4 + wave) * 16;
  const size_t base = (size_t)blockIdx.y * S * HD;
  const bf16* Qh = (const bf16*)Qv + base;
  const bf16* Kh = (const bf16*)Kv + base;
  const bf16* Vh = (const bf16*)Vv + base;
  bf16*       Oh = (bf16*)Ov + base;
  const bool causal = (maskp[0] != 0);

  // Q fragments: 16 rows x HD=64, two K-chunks of 32
  bf16x16 aq[2];
#pragma unroll
  for (int c = 0; c < 2; ++c) {
    const bf16* p = Qh + (size_t)(qbase + l16) * HD + c * 32 + hl * 8;
    aq[c] = ld_frag(p, p + 16);
  }

  f32x8 O[4];
#pragma unroll
  for (int n = 0; n < 4; ++n) O[n] = f32x8{};
  float m[8], l[8];
#pragma unroll
  for (int r = 0; r < 8; ++r) { m[r] = -1e30f; l[r] = 0.f; }

  bf16* pl = lP + wave * 512;

  // Fully-unmasked bulk: every key tile with kb+31 <= qbase needs no mask.
  const int kfull = causal ? (qbase & ~31) : S;
  for (int kb = 0; kb < kfull; kb += 32) {
    // prefetch next key/value tile rows (hint only; OOB-safe inside workspace)
    __builtin_prefetch(Kh + (size_t)(kb + 32 + lane) * HD, 0, 1);
    __builtin_prefetch(Vh + (size_t)(kb + 32 + lane) * HD, 0, 1);
    attn_tile<false>(kb, qbase, l16, hl, Kh, Vh, pl, aq, O, m, l);
  }
  // Exactly one diagonal-straddling tile when causal.
  if (causal) attn_tile<true>(kfull, qbase, l16, hl, Kh, Vh, pl, aq, O, m, l);

  // ---- normalize and write head-contiguous output (bf16)
#pragma unroll
  for (int n = 0; n < 4; ++n)
#pragma unroll
    for (int r = 0; r < 8; ++r) {
      const int row = qbase + r + hl * 8;
      Oh[(size_t)row * HD + n * 16 + l16] = f2b(O[n][r] / l[r]);
    }
}

// ---------------------------------------------------------------- launch
extern "C" void kernel_launch(void* const* d_in, const int* in_sizes, int n_in,
                              void* d_out, int out_size, void* d_ws, size_t ws_size,
                              hipStream_t stream) {
  const float* q  = (const float*)d_in[0];
  const float* k  = (const float*)d_in[1];
  const float* v  = (const float*)d_in[2];
  const float* wq = (const float*)d_in[3];
  const float* wk = (const float*)d_in[4];
  const float* wv = (const float*)d_in[5];
  const float* wo = (const float*)d_in[6];
  const int* mask = (const int*)d_in[7];
  (void)in_sizes; (void)n_in; (void)out_size; (void)ws_size;

  // Workspace: bf16 Qp/Kp/Vp/AttnOut, 8 MB each (32 MB total, L2-resident)
  const size_t P = (size_t)Mrows * E;
  unsigned short* qp = (unsigned short*)d_ws;
  unsigned short* kp = qp + P;
  unsigned short* vp = kp + P;
  unsigned short* ao = vp + P;

  const float scale = 0.125f;  // 1/sqrt(HD), folded into the Q projection
  dim3 gg(Mrows / 128, GN / 64);   // 32 x 16 blocks
  gemm_wmma<float, unsigned short><<<gg, 256, 0, stream>>>(q, wq, qp, scale);
  gemm_wmma<float, unsigned short><<<gg, 256, 0, stream>>>(k, wk, kp, 1.0f);
  gemm_wmma<float, unsigned short><<<gg, 256, 0, stream>>>(v, wv, vp, 1.0f);

  flash_attn<<<dim3(S / 64, Bsz * H), 128, 0, stream>>>(qp, kp, vp, ao, mask);

  gemm_wmma<unsigned short, float><<<gg, 256, 0, stream>>>(ao, wo, (float*)d_out, 1.0f);
}
